// DPIneuron_85633057948347
// MI455X (gfx1250) — matrix operand order repaired and checked by reading, not compiled, not run on hardware
//
#include <hip/hip_runtime.h>

// ---------------- CDNA5 WMMA / TDM types ----------------
typedef __attribute__((ext_vector_type(16))) _Float16     v16h;
typedef __attribute__((ext_vector_type(8)))  float        v8f;
typedef __attribute__((ext_vector_type(4)))  unsigned int v4u;
typedef __attribute__((ext_vector_type(8)))  int          v8i;
typedef __attribute__((ext_vector_type(4)))  int          v4i;

// ---------------- DPI neuron constants (precomputed in double, truncated) ----
// KAPPA = 0.705, UT = 25e-3, I0 = 0.5e-13
#define C_I0        5.0e-14f
#define C_ITAU      4.25e-12f
#define C_IGAIN     5.965e-11f
#define C_ALPHA     1.47e9f
#define C_P_I0      1.58058e-8f      // I0^(1/(kappa+1))
#define C_PKAP      0.41348973f      // kappa/(kappa+1)
#define C_RGT       14.035294f       // IGAIN/ITAU
#define C_INV_ITAU  2.3529412e11f    // 1/ITAU
#define C_DT_TAU    0.039950076f     // DT / TAU_MEM
#define C_F_DIS     0.93192776f      // exp(-DT / TAU_AMPA)
#define C_F_CHG     7.047515e-4f     // 1 - exp(-1e-5 / TAU_AMPA)
#define C_SYN_SCALE 2.0e-12f         // (IAMPA_G/IAMPA_TAU) * IAMPA_W0
#define C_SPIKE_TH  1.5e-4f

// One wave32 computes one 16x16 output tile:
//   isyn tile = input[16x128] @ W[16x128]^T  via 4x v_wmma_f32_16x16x32_f16
// The 16x128 W tile is shared by all 8 waves of the block -> staged once into
// LDS by the Tensor Data Mover (tensor_load_to_lds, TENSORcnt), then each wave
// builds its B fragments from LDS (ds_load_b128). A fragments are per-wave and
// read directly from the L2-resident input. DPI neuron elementwise update and
// spike store are fused behind the WMMA accumulator.
__global__ __launch_bounds__(256) void dpi_fused_kernel(
    const float* __restrict__ input,   // [B, 128]
    const float* __restrict__ W,       // [Nout, 128]
    const float* __restrict__ Imem,    // [B, Nout]
    const float* __restrict__ Iahp,
    const float* __restrict__ Tref,
    const float* __restrict__ Iampa,
    const float* __restrict__ Inmda,
    const float* __restrict__ Ishunt,
    const float* __restrict__ Igaba,
    float* __restrict__ spike,         // [B, Nout]
    float* __restrict__ iampa_out,     // workspace state update
    int Nout, int storeAmpa)
{
    __shared__ float sW[16][128];            // 8 KB W tile, f32

    const int lane  = threadIdx.x & 31;
    const int wave  = threadIdx.x >> 5;
    const int l15   = lane & 15;
    const int hi    = lane >> 4;             // 0: lanes 0-15, 1: lanes 16-31
    const int oTile = blockIdx.x;            // 16 output columns (shared W tile)
    const int bTile = blockIdx.y * 8 + wave; // 16 batch rows per wave

    // ---- TDM: DMA the 16x128 f32 W tile into LDS (one descriptor per block).
    if (wave == 0) {
        const unsigned long long ga =
            (unsigned long long)(W + (size_t)oTile * 16 * 128);
        const unsigned lds_off = (unsigned)(unsigned long long)(void*)&sW[0][0];

        v4u g0;
        g0[0] = 1u;                                   // count=1, user descriptor
        g0[1] = lds_off;                              // lds_addr (bytes)
        g0[2] = (unsigned)(ga & 0xFFFFFFFFu);         // global_addr[31:0]
        g0[3] = (unsigned)((ga >> 32) & 0x01FFFFFFu)  // global_addr[56:32]
                | (2u << 30);                         // type = 2 ("image")

        v8i g1;
        g1[0] = 0x00020000;                   // wg_mask=0 | data_size=2 (4B)
        g1[1] = (int)(128u << 16);            // tensor_dim0 = 128 (bits 79:48)
        g1[2] = (int)((unsigned)Nout << 16);  // tensor_dim1 = Nout (bits 111:80)
        g1[3] = (int)(128u << 16);            // tile_dim0 = 128 (bits 127:112)
        g1[4] = 16;                           // tile_dim1 = 16
        g1[5] = 128;                          // tensor_dim0_stride = 128
        g1[6] = 0;
        g1[7] = 0;

        v4i z4 = {0, 0, 0, 0};                // 2-D tile: groups 2/3 unused
#if defined(__clang_major__) && (__clang_major__ >= 23)
        v8i z8 = {0, 0, 0, 0, 0, 0, 0, 0};
        __builtin_amdgcn_tensor_load_to_lds(g0, g1, z4, z4, z8, 0);
#else
        __builtin_amdgcn_tensor_load_to_lds(g0, g1, z4, z4, 0);
#endif
        __builtin_amdgcn_s_wait_tensorcnt(0);
    }
    __syncthreads();

    // A fragment source row (M = l15) from global; B fragment from LDS tile.
    const float4* Arow4 = reinterpret_cast<const float4*>(
        input + (size_t)(bTile * 16 + l15) * 128);
    const float4* sWrow4 = reinterpret_cast<const float4*>(&sW[l15][0]);

    v8f acc = {};
    #pragma unroll
    for (int kb = 0; kb < 4; ++kb) {
        const int k0 = kb * 32;

        // ---- A fragment (16x32 f16): lane holds M=l15;
        //      elements 0-7  -> K = k0 + hi*8 + e      (contiguous 8)
        //      elements 8-15 -> K = k0 + 16 + hi*8 + e (contiguous 8)
        const int ka = (k0 + hi * 8) >> 2;
        const int kc = (k0 + 16 + hi * 8) >> 2;
        float4 a0 = Arow4[ka], a1 = Arow4[ka + 1];
        float4 a2 = Arow4[kc], a3 = Arow4[kc + 1];
        v16h af;
        af[0]  = (_Float16)a0.x; af[1]  = (_Float16)a0.y; af[2]  = (_Float16)a0.z; af[3]  = (_Float16)a0.w;
        af[4]  = (_Float16)a1.x; af[5]  = (_Float16)a1.y; af[6]  = (_Float16)a1.z; af[7]  = (_Float16)a1.w;
        af[8]  = (_Float16)a2.x; af[9]  = (_Float16)a2.y; af[10] = (_Float16)a2.z; af[11] = (_Float16)a2.w;
        af[12] = (_Float16)a3.x; af[13] = (_Float16)a3.y; af[14] = (_Float16)a3.z; af[15] = (_Float16)a3.w;

        // ---- B fragment (32x16 f16, column-striped): lane holds N=l15;
        //      elements 0-7 -> K = k0 + hi*8 + e, elements 8-15 -> K = k0 + 16 + hi*8 + e (from LDS)
        const int kbb = (k0 + hi * 8) >> 2;
        const int kbc = (k0 + 16 + hi * 8) >> 2;
        float4 b0 = sWrow4[kbb], b1 = sWrow4[kbb + 1];
        float4 b2 = sWrow4[kbc], b3 = sWrow4[kbc + 1];
        v16h bf;
        bf[0]  = (_Float16)b0.x; bf[1]  = (_Float16)b0.y; bf[2]  = (_Float16)b0.z; bf[3]  = (_Float16)b0.w;
        bf[4]  = (_Float16)b1.x; bf[5]  = (_Float16)b1.y; bf[6]  = (_Float16)b1.z; bf[7]  = (_Float16)b1.w;
        bf[8]  = (_Float16)b2.x; bf[9]  = (_Float16)b2.y; bf[10] = (_Float16)b2.z; bf[11] = (_Float16)b2.w;
        bf[12] = (_Float16)b3.x; bf[13] = (_Float16)b3.y; bf[14] = (_Float16)b3.z; bf[15] = (_Float16)b3.w;

        // (neg_a, A, neg_b, B, c_mod, C, reuse_a, reuse_b)
        acc = __builtin_amdgcn_wmma_f32_16x16x32_f16(false, af, false, bf,
                                                     (short)0, acc, false, false);
    }

    // ---- Fused DPI neuron update. C/D layout: VGPR r <-> row (r + 8*hi), col l15.
    const int ocol = oTile * 16 + l15;
    #pragma unroll
    for (int r = 0; r < 8; ++r) {
        const int    brow = bTile * 16 + hi * 8 + r;
        const size_t idx  = (size_t)brow * (size_t)Nout + (size_t)ocol;

        const float imem = Imem[idx];
        const float iahp = Iahp[idx];
        const float tr   = Tref[idx];
        const float iam  = Iampa[idx];
        const float inm  = Inmda[idx];
        const float ish  = Ishunt[idx];
        const float iga  = Igaba[idx];

        const float Ileak = C_ITAU + iahp + iga;
        float Iin = iam + inm - ish;                 // IDC == 0
        Iin = (tr <= 0.0f) ? Iin : 0.0f;
        Iin = fmaxf(Iin, C_I0);

        // positive feedback current
        const float Ifb = C_P_I0 * __powf(imem, C_PKAP)
                          / (1.0f + __expf(-C_ALPHA * (imem - C_IGAIN)));
        const float f_imem   = Ifb / Ileak * (imem + C_IGAIN);
        const float Imem_inf = C_RGT * (Iin - Ileak);
        const float dImem = imem / (imem + C_IGAIN)
                            * (Imem_inf + f_imem - imem * (1.0f + iahp * C_INV_ITAU))
                            * C_DT_TAU;
        const float Imem_new = fmaxf(imem + dImem, C_I0);
        spike[idx] = (Imem_new - C_SPIKE_TH > 0.0f) ? 1.0f : 0.0f;

        // AMPA state update (consumes the WMMA result) -> workspace
        const float isyn      = C_SYN_SCALE * acc[r];
        const float iampa_new = fmaxf(C_F_DIS * iam + C_F_CHG * isyn, C_I0);
        if (storeAmpa) iampa_out[idx] = iampa_new;
    }
}

extern "C" void kernel_launch(void* const* d_in, const int* in_sizes, int n_in,
                              void* d_out, int out_size, void* d_ws, size_t ws_size,
                              hipStream_t stream) {
    const float* input  = (const float*)d_in[0];
    const float* W      = (const float*)d_in[1];
    const float* Imem   = (const float*)d_in[2];
    const float* Iahp   = (const float*)d_in[3];
    const float* Tref   = (const float*)d_in[4];
    const float* Iampa  = (const float*)d_in[5];
    const float* Inmda  = (const float*)d_in[6];
    const float* Ishunt = (const float*)d_in[7];
    const float* Igaba  = (const float*)d_in[8];

    const int Bn   = in_sizes[0] / 128;   // 4096
    const int Nout = in_sizes[1] / 128;   // 4096
    const int storeAmpa =
        (ws_size >= (size_t)Bn * (size_t)Nout * sizeof(float)) ? 1 : 0;

    // 8 waves per block, one 16x16 tile per wave; block shares one W tile.
    dim3 grid((unsigned)(Nout / 16), (unsigned)(Bn / 128), 1);
    dpi_fused_kernel<<<grid, 256, 0, stream>>>(
        input, W, Imem, Iahp, Tref, Iampa, Inmda, Ishunt, Igaba,
        (float*)d_out, (float*)d_ws, Nout, storeAmpa);
}
